// ProgressiveVQVAE_41549513621531
// MI455X (gfx1250) — matrix operand order repaired and checked by reading, not compiled, not run on hardware
//
#include <hip/hip_runtime.h>
#include <cstdint>
#include <cstddef>

// ---------------------------------------------------------------------------
// MI455X (gfx1250) VQ-VAE forward, WMMA bf16 GEMM pipeline.
//
// Key design: NO LDS in the GEMM. Both operands live in global memory in
// WMMA-fragment-packed layout (1KB block per 16x32 / 32x16 tile, lane-major),
// so every fragment is two coalesced global_load_b128 per lane. Weights are
// <=3MB (L2-resident, 192MB L2); the A band is shared across N-blocks via L2.
// Producers (convert / LN+GELU) emit the packed layout with aligned uint4
// stores (8 consecutive k == one 16B fragment chunk).
// ---------------------------------------------------------------------------

typedef unsigned short u16;
typedef __attribute__((ext_vector_type(16))) __bf16 v16bf;
typedef __attribute__((ext_vector_type(8)))  float  v8f;

union FragU { v16bf v; uint4 q[2]; };

__device__ __forceinline__ u16 f32_bf16(float f) {
  unsigned x = __float_as_uint(f);
  x += 0x7FFFu + ((x >> 16) & 1u);  // round-to-nearest-even
  return (u16)(x >> 16);
}
__device__ __forceinline__ float bf16_f32(u16 h) {
  return __uint_as_float(((unsigned)h) << 16);
}
__device__ __forceinline__ float gelu_exact(float v) {
  return 0.5f * v * (1.0f + erff(v * 0.70710678118654752f));
}

// Packed-A address helpers (16m x 32k tile -> 32 lanes x 16 u16 = 1KB):
// lane = (m%16) + 16*hl, hl = (kk>>3)&1, e = (kk&7) + ((kk>>4)<<3), kk = k%32.
// For c0 multiple of 8: hl = (c0>>3)&1, e0 = ((c0&31)>>4)<<3 -> aligned 16B.
__device__ __forceinline__ u16* packA_ptr(u16* base, int row, int c0, int Kp) {
  const int kk = c0 & 31;
  const int hl = (kk >> 3) & 1;
  const int e0 = (kk >> 4) << 3;
  const int laneSlot = (row & 15) + (hl << 4);
  return base + ((size_t)(row >> 4) * (Kp >> 5) + (c0 >> 5)) * 512 + laneSlot * 16 + e0;
}

// ---------------------------------------------------------------------------
// f32 [R,srcC] -> packed-A bf16 [R, dstC] (cols >= srcC zero-padded).
// One thread = 8 consecutive cols = one aligned 16B packed chunk.
// ---------------------------------------------------------------------------
__global__ void conv_pack_a_kernel(const float* __restrict__ src, u16* __restrict__ dst,
                                   int srcC, int dstC, long nChunks) {
  long i = (long)blockIdx.x * 256 + threadIdx.x;
  if (i >= nChunks) return;
  const int cpr = dstC >> 3;
  const int r  = (int)(i / cpr);
  const int c0 = (int)(i % cpr) << 3;
  u16 h[8];
#pragma unroll
  for (int j = 0; j < 8; ++j) {
    const int c = c0 + j;
    h[j] = f32_bf16((c < srcC) ? src[(size_t)r * srcC + c] : 0.0f);
  }
  *(uint4*)packA_ptr(dst, r, c0, dstC) = *(const uint4*)h;
}

// ---------------------------------------------------------------------------
// Weights f32 [srcK,N] -> packed-B bf16 [dstK,N] (k >= srcK zero-padded).
// B tile (32k x 16n): lane = (n%16) + 16*((k>>3)&1), e = (k&7) + 8*((k>>4)&1).
// Thread = (n, 8 consecutive k) -> one aligned 16B chunk; reads coalesced
// across threads (n fastest).
// ---------------------------------------------------------------------------
__global__ void conv_pack_b_kernel(const float* __restrict__ src, u16* __restrict__ dst,
                                   int srcK, int N, int dstK, long nChunks) {
  long i = (long)blockIdx.x * 256 + threadIdx.x;
  if (i >= nChunks) return;
  const int n  = (int)(i % N);
  const int k0 = (int)(i / N) << 3;
  u16 h[8];
#pragma unroll
  for (int j = 0; j < 8; ++j) {
    const int k = k0 + j;
    h[j] = f32_bf16((k < srcK) ? src[(size_t)k * N + n] : 0.0f);
  }
  const int hl = (k0 >> 3) & 1;
  const int e0 = ((k0 >> 4) & 1) << 3;  // 0 or 8
  const int laneSlot = (n & 15) + (hl << 4);
  u16* p = dst + ((size_t)(n >> 4) * (dstK >> 5) + (k0 >> 5)) * 512 + laneSlot * 16 + e0;
  *(uint4*)p = *(const uint4*)h;
}

__global__ void zero_kernel(float* p, int n) {
  int i = blockIdx.x * 256 + threadIdx.x;
  if (i < n) p[i] = 0.0f;
}

// ---------------------------------------------------------------------------
// LDS-free bf16 WMMA GEMM: C[M,N] = A[M,K] @ B[K,N] + bias.
// Block = 256 threads = 8 waves (2x4); wave tile 64x32 = 8 accumulators.
// A/B fragments loaded straight from packed global layout, register
// double-buffered. Output row-major (f32 for z_e, bf16 otherwise).
// ---------------------------------------------------------------------------
#define GEMM_LOAD(AF, BF, kk)                                                \
  {                                                                          \
    const size_t koff = (size_t)(kk) * 512;                                  \
    _Pragma("unroll") for (int mi = 0; mi < 4; ++mi) {                       \
      AF[mi].q[0] = *(const uint4*)(aBase[mi] + koff);                       \
      AF[mi].q[1] = *(const uint4*)(aBase[mi] + koff + 8);                   \
    }                                                                        \
    _Pragma("unroll") for (int ni = 0; ni < 2; ++ni) {                       \
      BF[ni].q[0] = *(const uint4*)(bBase[ni] + koff);                       \
      BF[ni].q[1] = *(const uint4*)(bBase[ni] + koff + 8);                   \
    }                                                                        \
  }

#define GEMM_MMA(AF, BF)                                                     \
  _Pragma("unroll") for (int mi = 0; mi < 4; ++mi)                           \
  _Pragma("unroll") for (int ni = 0; ni < 2; ++ni)                           \
      acc[mi][ni] = __builtin_amdgcn_wmma_f32_16x16x32_bf16(                 \
          false, AF[mi].v, false, BF[ni].v, (short)0, acc[mi][ni], false, false);

__global__ __launch_bounds__(256) void gemm_bf16_kernel(
    const u16* __restrict__ Apk, const u16* __restrict__ Bpk,
    const float* __restrict__ bias, void* __restrict__ Cout,
    int N, int K, int outF32) {
  const int tid  = threadIdx.x;
  const int lane = tid & 31;
  const int wave = tid >> 5;
  const int wm   = wave >> 2;   // 0..1
  const int wn   = wave & 3;    // 0..3
  const int lr   = lane & 15;
  const int hl   = lane >> 4;
  const int bm   = blockIdx.y << 7;
  const int bn   = blockIdx.x << 7;
  const int kt   = K >> 5;

  const u16* aBase[4];
  const u16* bBase[2];
#pragma unroll
  for (int mi = 0; mi < 4; ++mi)
    aBase[mi] = Apk + (size_t)((bm >> 4) + (wm << 2) + mi) * kt * 512 + lane * 16;
#pragma unroll
  for (int ni = 0; ni < 2; ++ni)
    bBase[ni] = Bpk + (size_t)((bn >> 4) + (wn << 1) + ni) * kt * 512 + lane * 16;

  v8f acc[4][2] = {};
  FragU a0[4], b0[2], a1[4], b1[2];

  GEMM_LOAD(a0, b0, 0)
  int k = 0;
  while (k + 2 <= kt) {
    if (k + 8 < kt) __builtin_prefetch(aBase[0] + (size_t)(k + 8) * 512, 0, 1);
    GEMM_LOAD(a1, b1, k + 1)
    GEMM_MMA(a0, b0)
    if (k + 2 < kt) GEMM_LOAD(a0, b0, k + 2)
    GEMM_MMA(a1, b1)
    k += 2;
  }
  if (k < kt) GEMM_MMA(a0, b0)

  // ---- epilogue: bias + row-major store ----
#pragma unroll
  for (int mi = 0; mi < 4; ++mi) {
#pragma unroll
    for (int ni = 0; ni < 2; ++ni) {
      const int col = bn + wn * 32 + ni * 16 + lr;
      const float bv = bias ? bias[col] : 0.0f;
#pragma unroll
      for (int v = 0; v < 8; ++v) {
        const int row = bm + wm * 64 + mi * 16 + hl * 8 + v;
        const float val = acc[mi][ni][v] + bv;
        if (outF32) ((float*)Cout)[(size_t)row * N + col] = val;
        else        ((u16*)Cout)[(size_t)row * N + col]   = f32_bf16(val);
      }
    }
  }
}

// ---------------------------------------------------------------------------
// Rowwise LayerNorm (optional) + exact GELU on bf16.
// Input: row-major. Output: packed-A layout (outPK) OR row-major (outRM).
// One 256-thread block per row; one thread = one aligned 8-col chunk.
// ---------------------------------------------------------------------------
__global__ __launch_bounds__(256) void ln_gelu_kernel(
    const u16* __restrict__ in, u16* __restrict__ outRM, u16* __restrict__ outPK,
    const float* __restrict__ g, const float* __restrict__ b, int N, int doLN) {
  __shared__ float red[256], red2[256];
  __shared__ float s_mean, s_rstd;
  const int row = blockIdx.x;
  const int tid = threadIdx.x;
  const int active = tid < (N >> 3);
  const int c0 = tid << 3;

  float x[8];
  float sum = 0.f, sumsq = 0.f;
  if (active) {
    const uint4 raw = *(const uint4*)(in + (size_t)row * N + c0);
    const u16* hp = (const u16*)&raw;
#pragma unroll
    for (int j = 0; j < 8; ++j) {
      const float v = bf16_f32(hp[j]);
      x[j] = v;
      sum += v;
      sumsq += v * v;
    }
  }
  float m = 0.f, rs = 1.f;
  if (doLN) {
    red[tid] = sum;
    red2[tid] = sumsq;
    __syncthreads();
    for (int s = 128; s > 0; s >>= 1) {
      if (tid < s) { red[tid] += red[tid + s]; red2[tid] += red2[tid + s]; }
      __syncthreads();
    }
    if (tid == 0) {
      const float mean = red[0] / N;
      const float var  = red2[0] / N - mean * mean;
      s_mean = mean;
      s_rstd = rsqrtf(var + 1e-5f);
    }
    __syncthreads();
    m = s_mean;
    rs = s_rstd;
  }
  if (active) {
    u16 h[8];
#pragma unroll
    for (int j = 0; j < 8; ++j) {
      float v = x[j];
      if (doLN) v = (v - m) * rs * g[c0 + j] + b[c0 + j];
      h[j] = f32_bf16(gelu_exact(v));
    }
    if (outPK) *(uint4*)packA_ptr(outPK, row, c0, N) = *(const uint4*)h;
    else       *(uint4*)(outRM + (size_t)row * N + c0) = *(const uint4*)h;
  }
}

// ---------------------------------------------------------------------------
// VQ: argmin over 256 codes, z_q gather, commitment-loss acc, histogram.
// 16 rows/block; codebook tiled 64-at-a-time into LDS, per-code rotation.
// ---------------------------------------------------------------------------
__global__ __launch_bounds__(256) void vq_kernel(
    const float* __restrict__ zE, const float* __restrict__ cb,
    float* __restrict__ zQ, float* __restrict__ idxOut,
    float* __restrict__ lossAcc, float* __restrict__ counts) {
  __shared__ float sZ[16 * 128];
  __shared__ float sCB[64 * 128];
  __shared__ float sBest[256];
  __shared__ int   sBestI[256];
  __shared__ int   sIdx[16];
  __shared__ float sLoss[256];

  const int tid = threadIdx.x;
  const int r   = tid >> 4;
  const int sc  = tid & 15;
  const size_t rowBase = (size_t)blockIdx.x * 16;

  for (int i = tid; i < 16 * 128; i += 256) sZ[i] = zE[rowBase * 128 + i];

  float best = 3.4e38f;
  int bestI = 0;
  for (int t = 0; t < 4; ++t) {
    __syncthreads();
    for (int i = tid; i < 64 * 128; i += 256) {
      const int c = i >> 7, d = i & 127;
      sCB[(c << 7) + ((d + c) & 127)] = cb[(size_t)((t << 6) + c) * 128 + d];
    }
    __syncthreads();
#pragma unroll
    for (int j = 0; j < 4; ++j) {
      const int c = (sc << 2) + j;
      const float* zr = &sZ[r << 7];
      const float* cr = &sCB[c << 7];
      float acc = 0.f;
#pragma unroll 8
      for (int d = 0; d < 128; ++d) {
        const float diff = zr[d] - cr[(d + c) & 127];
        acc = fmaf(diff, diff, acc);
      }
      const int code = (t << 6) + c;
      if (acc < best) { best = acc; bestI = code; }
    }
  }
  sBest[tid] = best;
  sBestI[tid] = bestI;
  __syncthreads();
  for (int s = 8; s > 0; s >>= 1) {
    if (sc < s) {
      if (sBest[tid + s] < sBest[tid]) {
        sBest[tid]  = sBest[tid + s];
        sBestI[tid] = sBestI[tid + s];
      }
    }
    __syncthreads();
  }
  if (sc == 0) {
    const int bi = sBestI[r * 16];
    sIdx[r] = bi;
    idxOut[rowBase + r] = (float)bi;
    atomicAdd(&counts[bi], 1.0f);
  }
  __syncthreads();

  float ls = 0.f;
  {
    const int bi = sIdx[r];
    const float* crow = cb + (size_t)bi * 128;
#pragma unroll
    for (int kk = 0; kk < 8; ++kk) {
      const int d = sc * 8 + kk;
      const float q = crow[d];
      const float diff = q - sZ[(r << 7) + d];
      ls = fmaf(diff, diff, ls);
      zQ[(rowBase + r) * 128 + d] = q;
    }
  }
  sLoss[tid] = ls;
  __syncthreads();
  for (int s = 128; s > 0; s >>= 1) {
    if (tid < s) sLoss[tid] += sLoss[tid + s];
    __syncthreads();
  }
  if (tid == 0) atomicAdd(lossAcc, sLoss[0]);
}

// ---------------------------------------------------------------------------
// Head: [B,128] bf16 (row-major) @ D3[128,2] + db3 -> f32.
// ---------------------------------------------------------------------------
__global__ __launch_bounds__(256) void head_kernel(const u16* __restrict__ din,
                                                   const float* __restrict__ W,
                                                   const float* __restrict__ b,
                                                   float* __restrict__ out, int Bn) {
  const int r = blockIdx.x * 256 + threadIdx.x;
  if (r >= Bn) return;
  const u16* p = din + (size_t)r * 128;
  float a0 = b[0], a1 = b[1];
#pragma unroll 4
  for (int d = 0; d < 128; ++d) {
    const float v = bf16_f32(p[d]);
    a0 = fmaf(v, W[d * 2 + 0], a0);
    a1 = fmaf(v, W[d * 2 + 1], a1);
  }
  out[(size_t)r * 2 + 0] = a0;
  out[(size_t)r * 2 + 1] = a1;
}

__global__ void finalize_kernel(const float* __restrict__ counts,
                                const float* __restrict__ lossAcc,
                                float* __restrict__ outLoss, float* __restrict__ outPerp,
                                float invB, float invBM) {
  __shared__ float red[256];
  const int t = threadIdx.x;
  const float p = counts[t] * invB;
  red[t] = -p * logf(p + 1e-10f);
  __syncthreads();
  for (int s = 128; s > 0; s >>= 1) {
    if (t < s) red[t] += red[t + s];
    __syncthreads();
  }
  if (t == 0) {
    *outPerp = expf(red[0]);
    *outLoss = 0.1f * lossAcc[0] * invBM;
  }
}

// ---------------------------------------------------------------------------
// Host launcher. Workspace peak ~330 MiB (regions aliased; see comments).
// ---------------------------------------------------------------------------
extern "C" void kernel_launch(void* const* d_in, const int* in_sizes, int n_in,
                              void* d_out, int out_size, void* d_ws, size_t ws_size,
                              hipStream_t stream) {
  (void)in_sizes; (void)n_in; (void)out_size; (void)ws_size;
  const int B = 65536;

  const float* x   = (const float*)d_in[0];
  const float* W1  = (const float*)d_in[1];  const float* b1  = (const float*)d_in[2];
  const float* g1  = (const float*)d_in[3];  const float* be1 = (const float*)d_in[4];
  const float* W2  = (const float*)d_in[5];  const float* b2  = (const float*)d_in[6];
  const float* g2  = (const float*)d_in[7];  const float* be2 = (const float*)d_in[8];
  const float* W3  = (const float*)d_in[9];  const float* b3  = (const float*)d_in[10];
  const float* g3  = (const float*)d_in[11]; const float* be3 = (const float*)d_in[12];
  const float* W4  = (const float*)d_in[13]; const float* b4  = (const float*)d_in[14];
  const float* CB  = (const float*)d_in[15];
  const float* D1  = (const float*)d_in[16]; const float* db1 = (const float*)d_in[17];
  const float* dg1 = (const float*)d_in[18]; const float* dbe1= (const float*)d_in[19];
  const float* D2  = (const float*)d_in[20]; const float* db2 = (const float*)d_in[21];
  const float* D3  = (const float*)d_in[22]; const float* db3 = (const float*)d_in[23];

  float* out  = (float*)d_out;
  float* vel  = out;                    // [B,2]
  float* zE   = out + (size_t)2 * B;    // [B,128] f32
  float* zQ   = out + (size_t)130 * B;  // [B,128] f32
  float* idxO = out + (size_t)258 * B;  // [B]
  float* lossO= out + (size_t)259 * B;
  float* perpO= lossO + 1;

  const size_t MiB = 1ull << 20;
  char* wsb = (char*)d_ws;
  // Region X (base 0, 180 MiB): xp dead after GEMM1.
  u16* xp    = (u16*)(wsb + 0);          // [B,1440] packed, 180 MiB
  u16* act1p = (u16*)(wsb + 0);          // [B,1024] packed, 128 MiB
  u16* act3r = (u16*)(wsb + 136 * MiB);  // [B,256] row-major, 32 MiB
  u16* zqp   = (u16*)(wsb + 0);          // [B,128] packed, 16 MiB
  u16* dec1p = (u16*)(wsb + 24 * MiB);   // [B,256] packed, 32 MiB
  // Region Y (base 184 MiB, 136 MiB):
  u16* act1r = (u16*)(wsb + 184 * MiB);          // [B,1024] row-major, 128 MiB
  u16* act2r = (u16*)(wsb + 184 * MiB);          // [B,512] row-major, 64 MiB
  u16* act2p = (u16*)(wsb + (184 + 72) * MiB);   // [B,512] packed, 64 MiB
  u16* act3p = (u16*)(wsb + 184 * MiB);          // [B,256] packed, 32 MiB
  u16* dec1r = (u16*)(wsb + (184 + 40) * MiB);   // [B,256] row-major, 32 MiB
  u16* dec2r = (u16*)(wsb + (184 + 80) * MiB);   // [B,128] row-major, 16 MiB
  // Region Z (base 324 MiB): packed weights + stats.
  char* z = wsb + 324 * MiB;
  u16* w1p = (u16*)z; z += (size_t)1440 * 1024 * 2;
  u16* w2p = (u16*)z; z += (size_t)1024 * 512 * 2;
  u16* w3p = (u16*)z; z += (size_t)512 * 256 * 2;
  u16* w4p = (u16*)z; z += (size_t)256 * 128 * 2;
  u16* wD1p= (u16*)z; z += (size_t)128 * 256 * 2;
  u16* wD2p= (u16*)z; z += (size_t)256 * 128 * 2;
  float* stats   = (float*)z;
  float* counts  = stats;        // [256]
  float* lossAcc = stats + 256;  // [1]

  auto convA = [&](const float* s, u16* d, long R, int srcC, int dstC) {
    const long chunks = R * (dstC >> 3);
    conv_pack_a_kernel<<<(int)((chunks + 255) / 256), 256, 0, stream>>>(s, d, srcC, dstC, chunks);
  };
  auto convB = [&](const float* s, u16* d, int srcK, int N, int dstK) {
    const long chunks = (long)N * (dstK >> 3);
    conv_pack_b_kernel<<<(int)((chunks + 255) / 256), 256, 0, stream>>>(s, d, srcK, N, dstK, chunks);
  };
  auto gemm = [&](const u16* A, const u16* Bw, const float* bias, void* C,
                  int N, int K, int f32out) {
    dim3 grid(N / 128, B / 128);
    gemm_bf16_kernel<<<grid, 256, 0, stream>>>(A, Bw, bias, C, N, K, f32out);
  };

  // ---- pack inputs & weights to bf16 (K padded 1420 -> 1440 for layer 1) ----
  convA(x, xp, B, 1420, 1440);
  convB(W1, w1p, 1420, 1024, 1440);
  convB(W2, w2p, 1024, 512, 1024);
  convB(W3, w3p, 512, 256, 512);
  convB(W4, w4p, 256, 128, 256);
  convB(D1, wD1p, 128, 256, 128);
  convB(D2, wD2p, 256, 128, 256);
  zero_kernel<<<2, 256, 0, stream>>>(stats, 257);

  // ---- encoder ----
  gemm(xp, w1p, b1, act1r, 1024, 1440, 0);
  ln_gelu_kernel<<<B, 256, 0, stream>>>(act1r, nullptr, act1p, g1, be1, 1024, 1);
  gemm(act1p, w2p, b2, act2r, 512, 1024, 0);
  ln_gelu_kernel<<<B, 256, 0, stream>>>(act2r, nullptr, act2p, g2, be2, 512, 1);
  gemm(act2p, w3p, b3, act3r, 256, 512, 0);
  ln_gelu_kernel<<<B, 256, 0, stream>>>(act3r, nullptr, act3p, g3, be3, 256, 1);
  gemm(act3p, w4p, b4, zE, 128, 256, 1);  // z_e f32 -> d_out

  // ---- vector quantization ----
  vq_kernel<<<B / 16, 256, 0, stream>>>(zE, CB, zQ, idxO, lossAcc, counts);
  convA(zQ, zqp, B, 128, 128);

  // ---- decoder ----
  gemm(zqp, wD1p, db1, dec1r, 256, 128, 0);
  ln_gelu_kernel<<<B, 256, 0, stream>>>(dec1r, nullptr, dec1p, dg1, dbe1, 256, 1);
  gemm(dec1p, wD2p, db2, dec2r, 128, 256, 0);
  ln_gelu_kernel<<<B, 256, 0, stream>>>(dec2r, dec2r, nullptr, nullptr, nullptr, 128, 0);
  head_kernel<<<(B + 255) / 256, 256, 0, stream>>>(dec2r, D3, db3, vel, B);

  // ---- scalars ----
  finalize_kernel<<<1, 256, 0, stream>>>(counts, lossAcc, lossO, perpO,
                                         1.0f / (float)B, 1.0f / ((float)B * 128.0f));
}
